// BayesianEncoder_72164040507893
// MI455X (gfx1250) — matrix operand order repaired and checked by reading, not compiled, not run on hardware
//
#include <hip/hip_runtime.h>
#include <hip/hip_bf16.h>

// ---------------------------------------------------------------------------
// BayesianEncoder on MI455X (gfx1250, wave32)
//
// Stage A (mlp_kernel, x2): 4-layer MLP via v_wmma_f32_16x16x32_f16.
//   Fragment loads are vectorized per the ISA VGPR layouts:
//     A (16x32 f16):  half j<8 -> K=kbase+8*hi+j ; j>=8 -> K=kbase+16+8*hi+(j-8)
//                     => two contiguous 16B chunks => 2x ds_load_b128
//     B (32x16 f16):  half j -> same K as A (two 8-K chunks) with weights
//                     staged TRANSPOSED in LDS (Wt[n][k]) => 2x ds_load_b128
// Stage B (cavi_kernel): all CAVI iterations in one kernel, one workgroup per
//   batch; r+s working set (134 MB f32) is L2-resident (192 MB). float4 loads.
// ---------------------------------------------------------------------------

typedef _Float16 f16;
typedef __attribute__((ext_vector_type(16))) _Float16 v16h;
typedef __attribute__((ext_vector_type(8)))  _Float16 v8h;
typedef __attribute__((ext_vector_type(2)))  _Float16 v2h;
typedef __attribute__((ext_vector_type(8)))  float    v8f;

#define HDIM 128
#define LDIM 128
#define BB   64
#define NN   2048
#define MROWS (BB * NN)   // 131072 points

union Frag { v16h v; v8h p[2]; };

// ---------------------------------------------------------------------------
// MLP kernel: 128 threads (4 waves), each wave computes a 16-row x 128-col
// tile of every layer. grid.x = MROWS/64 = 2048 blocks.
// ---------------------------------------------------------------------------
__global__ __launch_bounds__(128)
void mlp_kernel(const float* __restrict__ xc,  const float* __restrict__ yc,
                const float* __restrict__ Win, const float* __restrict__ bin,
                const float* __restrict__ W1,  const float* __restrict__ b1,
                const float* __restrict__ W2,  const float* __restrict__ b2,
                const float* __restrict__ Wout,const float* __restrict__ bout,
                float* __restrict__ out, int is_var)
{
    __shared__ f16 Wt[HDIM][HDIM];        // 32 KB, transposed weights: Wt[n][k]
    __shared__ f16 act[4][16][HDIM];      // 16 KB, per-wave activation tiles

    const int tid  = threadIdx.x;
    const int w    = tid >> 5;            // wave id 0..3
    const int lane = tid & 31;
    const int m    = lane & 15;           // A: row in tile / B: col n offset
    const int hi   = lane >> 4;           // half-wave selector
    const int rowBase = blockIdx.x * 64;  // 4 waves * 16 rows

    // ---- input layer: K = DX+DY = 2, plain VALU ----
    {
        const int l  = tid;               // blockDim == 128
        const float w0 = Win[l];
        const float w1 = Win[HDIM + l];
        const float bb = bin[l];
        for (int rloc = 0; rloc < 64; ++rloc) {
            const int M = rowBase + rloc; // xc,yc are [B,N,1] -> flat index M
            float h = fmaf(xc[M], w0, fmaf(yc[M], w1, bb));
            act[rloc >> 4][rloc & 15][l] = (f16)fmaxf(h, 0.0f);
        }
    }
    __syncthreads();

    const float* Ws[3] = { W1, W2, Wout };
    const float* bs[3] = { b1, b2, bout };

    for (int layer = 0; layer < 3; ++layer) {
        // stage transposed weights: Wt[n][k] = W[k*H + n], packed f16 pairs.
        // Global reads coalesced (consecutive tid -> consecutive n); LDS store
        // is one b32 per pair (k even -> 4B aligned).
        {
            const float* W = Ws[layer];
            for (int idx = tid; idx < HDIM * HDIM / 2; idx += 128) {
                const int n = idx & 127;
                const int k = (idx >> 7) << 1;
                v2h pk;
                pk[0] = (f16)W[k * HDIM + n];
                pk[1] = (f16)W[(k + 1) * HDIM + n];
                *(v2h*)&Wt[n][k] = pk;
            }
        }
        __syncthreads();

        v8f c[8];
        #pragma unroll
        for (int t = 0; t < 8; ++t)
            #pragma unroll
            for (int e = 0; e < 8; ++e) c[t][e] = 0.0f;

        #pragma unroll
        for (int ks = 0; ks < 4; ++ks) {
            const int kbase = ks * 32;
            // A fragment: two contiguous 16B chunks of this lane's row m
            Frag a;
            a.p[0] = *(const v8h*)&act[w][m][kbase + (hi << 3)];
            a.p[1] = *(const v8h*)&act[w][m][kbase + 16 + (hi << 3)];
            #pragma unroll
            for (int t = 0; t < 8; ++t) {
                // B fragment: two 8-K chunks of column n (transposed W), same K order as A
                const int n = (t << 4) + m;
                Frag bf;
                const f16* col = &Wt[n][kbase + (hi << 3)];
                bf.p[0] = *(const v8h*)&col[0];
                bf.p[1] = *(const v8h*)&col[16];
                c[t] = __builtin_amdgcn_wmma_f32_16x16x32_f16(
                           false, a.v, false, bf.v, (short)0, c[t], false, false);
            }
        }
        __syncthreads();   // all waves finished reading Wt

        const float* bias = bs[layer];
        if (layer < 2) {
            // bias + ReLU back into this wave's own activation tile
            #pragma unroll
            for (int t = 0; t < 8; ++t) {
                const int n = (t << 4) + m;
                const float bb = bias[n];
                #pragma unroll
                for (int g = 0; g < 8; ++g) {
                    const int mr = g + (hi << 3);      // C layout: M = g + 8*hi
                    act[w][mr][n] = (f16)fmaxf(c[t][g] + bb, 0.0f);
                }
            }
        } else {
            // output layer epilogue -> global (coalesced across lanes per g,t)
            #pragma unroll
            for (int t = 0; t < 8; ++t) {
                const int n = (t << 4) + m;
                const float bb = bias[n];
                #pragma unroll
                for (int g = 0; g < 8; ++g) {
                    const int mr = g + (hi << 3);
                    float v = c[t][g] + bb;
                    if (is_var) {
                        const float sg  = 1.0f / (1.0f + __expf(-v));
                        const float var = 1e-4f + (1.0f - 1e-4f) * sg;
                        v = 1.0f / var;                // store precision s = 1/var
                    }
                    const size_t M = (size_t)(rowBase + w * 16 + mr);
                    out[M * LDIM + n] = v;
                }
            }
        }
    }
}

// ---------------------------------------------------------------------------
// CAVI kernel: one workgroup per batch b, 1024 threads (32 waves), all
// iterations fused; float4 global traffic (16B-aligned everywhere).
// ---------------------------------------------------------------------------
__global__ __launch_bounds__(1024)
void cavi_kernel(const float* __restrict__ r_ws, const float* __restrict__ s_ws,
                 const int* __restrict__ max_steps, float* __restrict__ out)
{
    __shared__ float g[NN];                // gm_mom = a/b per n (8 KB)
    __shared__ float mu[LDIM], cov[LDIM];
    __shared__ float S1buf[32 * LDIM];     // partials: 32 n-groups x 128 l (16KB)
    __shared__ float S2buf[32 * LDIM];     // 16 KB
    __shared__ float dbuf[LDIM];           // mu^2 + cov per l (for d_new)
    __shared__ float gp;                   // gm_mom_prior = c/d

    const int b   = blockIdx.x;
    const int tid = threadIdx.x;
    const size_t baseB = (size_t)b * NN * LDIM;

    for (int n = tid; n < NN; n += 1024) g[n] = 1.0f;  // a=b=1
    if (tid == 0) gp = 1.0f;                           // c=d=1
    __syncthreads();

    const int steps = max_steps[0];
    const int total = (steps > 1) ? steps : 1;         // 1 + max(0, steps-1)

    const float gam_r     = 0.01f * LDIM;              // 1.28
    const float gam_prior = 1e-6f * LDIM;              // 1.28e-4
    const float a_new     = gam_r + 0.5f * LDIM;       // 65.28  (constant)
    const float c_new     = gam_prior + 0.5f * LDIM;   // 64.000128 (constant)

    for (int it = 0; it < total; ++it) {
        // ---------------- cavi_gaussian ----------------
        {
            const int l4  = (tid & 31) << 2;  // this lane's 4 contiguous l's
            const int grp = tid >> 5;         // 32 groups striding N
            float4 S1 = make_float4(0.f, 0.f, 0.f, 0.f);
            float4 S2 = make_float4(0.f, 0.f, 0.f, 0.f);
            for (int n = grp; n < NN; n += 32) {
                const float  gv  = g[n];
                const size_t idx = baseB + (size_t)n * LDIM + l4;
                const float4 sv = *(const float4*)&s_ws[idx];   // 1/var_r
                const float4 rv = *(const float4*)&r_ws[idx];   // r (mu_z==0)
                S1.x = fmaf(gv, sv.x, S1.x);  S2.x = fmaf(gv * sv.x, rv.x, S2.x);
                S1.y = fmaf(gv, sv.y, S1.y);  S2.y = fmaf(gv * sv.y, rv.y, S2.y);
                S1.z = fmaf(gv, sv.z, S1.z);  S2.z = fmaf(gv * sv.z, rv.z, S2.z);
                S1.w = fmaf(gv, sv.w, S1.w);  S2.w = fmaf(gv * sv.w, rv.w, S2.w);
            }
            *(float4*)&S1buf[grp * LDIM + l4] = S1;
            *(float4*)&S2buf[grp * LDIM + l4] = S2;
        }
        __syncthreads();
        if (tid < LDIM) {
            float S1 = 0.0f, S2 = 0.0f;
            #pragma unroll
            for (int grp = 0; grp < 32; ++grp) {
                S1 += S1buf[grp * LDIM + tid];
                S2 += S2buf[grp * LDIM + tid];
            }
            const float cz = 1.0f / (gp + S1);
            const float m  = cz * S2;
            cov[tid]  = cz;
            mu[tid]   = m;
            dbuf[tid] = m * m + cz;        // for d_new reduction
        }
        __syncthreads();

        if (it + 1 < total) {
            // ---------------- cavi_gamma ----------------
            if (tid == 0) {
                float ds = 0.0f;
                for (int l = 0; l < LDIM; ++l) ds += dbuf[l];
                const float d = gam_prior + 0.5f * ds;
                gp = c_new / d;            // prior precision for next gaussian
            }
            for (int n = tid; n < NN; n += 1024) {
                const size_t base = baseB + (size_t)n * LDIM;
                float rate = 0.0f;
                for (int l = 0; l < LDIM; l += 4) {
                    const float4 rv = *(const float4*)&r_ws[base + l];
                    const float4 sv = *(const float4*)&s_ws[base + l];
                    float dm;
                    dm = rv.x - mu[l + 0]; rate = fmaf(sv.x, dm * dm + cov[l + 0], rate);
                    dm = rv.y - mu[l + 1]; rate = fmaf(sv.y, dm * dm + cov[l + 1], rate);
                    dm = rv.z - mu[l + 2]; rate = fmaf(sv.z, dm * dm + cov[l + 2], rate);
                    dm = rv.w - mu[l + 3]; rate = fmaf(sv.w, dm * dm + cov[l + 3], rate);
                }
                g[n] = a_new / (gam_r + 0.5f * rate);   // a_new / b_new
            }
            __syncthreads();
        }
    }

    if (tid < LDIM) {
        out[(size_t)b * LDIM + tid] = mu[tid];                         // mu  [B,L]
        out[(size_t)(BB * LDIM) + (size_t)b * LDIM + tid] = cov[tid];  // cov [B,L]
    }
}

// ---------------------------------------------------------------------------
extern "C" void kernel_launch(void* const* d_in, const int* in_sizes, int n_in,
                              void* d_out, int out_size, void* d_ws, size_t ws_size,
                              hipStream_t stream)
{
    const float* xc     = (const float*)d_in[0];
    const float* yc     = (const float*)d_in[1];
    const float* r_Win  = (const float*)d_in[2];
    const float* r_bin  = (const float*)d_in[3];
    const float* r_W1   = (const float*)d_in[4];
    const float* r_b1   = (const float*)d_in[5];
    const float* r_W2   = (const float*)d_in[6];
    const float* r_b2   = (const float*)d_in[7];
    const float* r_Wout = (const float*)d_in[8];
    const float* r_bout = (const float*)d_in[9];
    const float* v_Win  = (const float*)d_in[10];
    const float* v_bin  = (const float*)d_in[11];
    const float* v_W1   = (const float*)d_in[12];
    const float* v_b1   = (const float*)d_in[13];
    const float* v_W2   = (const float*)d_in[14];
    const float* v_b2   = (const float*)d_in[15];
    const float* v_Wout = (const float*)d_in[16];
    const float* v_bout = (const float*)d_in[17];
    const int*   mxs    = (const int*)d_in[18];
    float*       outp   = (float*)d_out;

    const size_t elems = (size_t)MROWS * LDIM;          // 16,777,216
    const size_t need  = 2 * elems * sizeof(float);     // 134,217,728 B
    if (ws_size < need) return;                         // insufficient scratch

    float* r_ws = (float*)d_ws;
    float* s_ws = r_ws + elems;

    // Stage A: two WMMA MLP passes (r latents, then precision s = 1/var_r)
    mlp_kernel<<<MROWS / 64, 128, 0, stream>>>(
        xc, yc, r_Win, r_bin, r_W1, r_b1, r_W2, r_b2, r_Wout, r_bout, r_ws, 0);
    mlp_kernel<<<MROWS / 64, 128, 0, stream>>>(
        xc, yc, v_Win, v_bin, v_W1, v_b1, v_W2, v_b2, v_Wout, v_bout, s_ws, 1);

    // Stage B: fused 5-iteration CAVI, one workgroup per batch, L2-resident
    cavi_kernel<<<BB, 1024, 0, stream>>>(r_ws, s_ws, mxs, outp);
}
